// Node_65618510348951
// MI455X (gfx1250) — matrix operand order repaired and checked by reading, not compiled, not run on hardware
//
#include <hip/hip_runtime.h>
#include <hip/hip_bf16.h>
#include <math.h>

typedef __attribute__((ext_vector_type(16))) _Float16 v16h;
typedef __attribute__((ext_vector_type(8)))  _Float16 v8h;
typedef __attribute__((ext_vector_type(4)))  _Float16 v4h;
typedef __attribute__((ext_vector_type(8)))  float    v8f;

#define DEV static __device__ __forceinline__

// ---------------- problem constants ----------------
constexpr int BATCH = 262144;
constexpr int OUTW  = 324;      // mu16 var16 msgs128 h2_64 h50 c50

// ---------------- packed-weight layout (halves) ----------------
constexpr int FRAG = 512;                       // 16x16x32-f16 B fragment = 512 halves = 1KB
constexpr int F_Win    = 0;                     // W_in:   N=4  K=1  -> 4 frags
constexpr int F_msg1   = F_Win  + 4*FRAG;       // msg_W1: per port N=7 K=2 -> 14 frags
constexpr int F_msg1_s = 14*FRAG;
constexpr int F_msg2   = F_msg1 + 4*F_msg1_s;   // msg_W2: per port N=2 K=4 -> 8 frags
constexpr int F_msg2_s = 8*FRAG;
constexpr int F_ih     = F_msg2 + 4*F_msg2_s;   // W_ih:   N=16 K=6 -> 96 frags (gate-padded rows)
constexpr int F_hh     = F_ih   + 96*FRAG;      // W_hh:   N=16 K=2 -> 32 frags (gate-padded rows)
constexpr int F_u1     = F_hh   + 32*FRAG;      // W_u1:   N=7  K=2 -> 14 frags
constexpr int F_u2     = F_u1   + 14*FRAG;      // W_u2:   N=7  K=4 -> 28 frags
constexpr int F_u3     = F_u2   + 28*FRAG;      // W_u3:   N=4  K=4 -> 16 frags
constexpr int F_mv     = F_u3   + 16*FRAG;      // W_mu (N=1 K=2) then W_var (N=1 K=2) -> 4 frags
constexpr int F_END    = F_mv   + 4*FRAG;       // 144384 halves = 288768 B
// padded biases (floats) appended after fragments
constexpr int BI_in=0, BI_m1=64, BI_m2=512, BI_l=640, BI_u1=896, BI_u2=1008, BI_u3=1120, BI_mv=1184;

// ---------------- pack kernels (tiny; run each launch, deterministic) ----------------
// B-fragment layout (ISA 7.12.2, 16-bit B 32x16): lane L, half j -> col = L%16, K = j<8 ? 8*(L/16)+j : 16+8*(L/16)+(j-8)
__global__ void pack_frags(const float* __restrict__ W, _Float16* __restrict__ dst,
                           int nT, int kT, int outActual, int inActual, int gateMode) {
  int tid = blockIdx.x * 256 + threadIdx.x;
  int total = nT * kT * 512;
  if (tid >= total) return;
  int frag = tid >> 9, within = tid & 511;
  int lane = within >> 4, j = within & 15;
  int n = frag / kT, k = frag % kT;
  int col = n * 16 + (lane & 15);
  int kk  = k * 32 + ((j < 8) ? ((lane >> 4) << 3) + j : 16 + ((lane >> 4) << 3) + (j - 8));
  int row = col;
  if (gateMode) { int g = col >> 6, jj = col & 63; row = (jj < 50) ? g * 50 + jj : -1; }
  else if (col >= outActual) row = -1;
  float v = 0.f;
  if (row >= 0 && kk < inActual) v = W[(size_t)row * inActual + kk];
  dst[(size_t)frag * 512 + lane * 16 + j] = (_Float16)v;
}

__global__ void pack_bias(const float* __restrict__ src, float* __restrict__ dst, int n, int npad) {
  int i = blockIdx.x * 64 + threadIdx.x;
  if (i < npad) dst[i] = (i < n) ? src[i] : 0.f;
}

__global__ void pack_bias_lstm(const float* __restrict__ bih, const float* __restrict__ bhh,
                               float* __restrict__ dst) {
  int i = blockIdx.x * 64 + threadIdx.x;
  if (i >= 256) return;
  int g = i >> 6, jj = i & 63;
  dst[i] = (jj < 50) ? bih[g * 50 + jj] + bhh[g * 50 + jj] : 0.f;
}

// ---------------- CDNA5 async global->LDS (ASYNCcnt, ISA ch.10 / 08_async_tensor) ----------------
DEV void async_ld_b128(unsigned int lds_byte_off, const void* gptr) {
  // GV mode: LDS address taken from the VDST VGPR; global addr from 64-bit VGPR pair
  asm volatile("global_load_async_to_lds_b128 %0, %1, off"
               :: "v"(lds_byte_off), "v"((unsigned long long)(uintptr_t)gptr)
               : "memory");
}
DEV void wait_async0() { asm volatile("s_wait_asynccnt 0" ::: "memory"); }

// ---------------- fragment helpers ----------------
DEV v8f wmma16(v16h a, v16h b, v8f c) {
  // v_wmma_f32_16x16x32_f16: D = A(16x32) x B(32x16) + C
  return __builtin_amdgcn_wmma_f32_16x16x32_f16(false, a, false, b, (short)0, c, false, false);
}

DEV v16h loadB(const _Float16* __restrict__ frag, int lane) {
  return *(const v16h*)(frag + lane * 16);    // 32B contiguous per lane (two b128 loads)
}

// A layout (ISA 7.12.2, 16-bit A 16x32): lane L row = L%16;
// halves j<8 -> K = cb + 8*(L/16) + j ; halves j>=8 -> K = cb + 16 + 8*(L/16) + (j-8)
DEV v16h loadA(const _Float16* __restrict__ buf, int SW, int cb, int lane) {
  int row = lane & 15, hi = lane >> 4;
  const _Float16* p0 = buf + row * SW + cb + hi * 8;
  v8h lo = *(const v8h*)p0;
  v8h hh = *(const v8h*)(p0 + 16);
  v16h a;
#pragma unroll
  for (int j = 0; j < 8; ++j) { a[j] = lo[j]; a[8 + j] = hh[j]; }
  return a;
}

DEV v16h loadA_obs(const float* __restrict__ orow, int lane) {  // 16x32 tile straight from global
  int hi = lane >> 4;
  float4 f0 = *(const float4*)(orow + hi * 8);
  float4 f1 = *(const float4*)(orow + hi * 8 + 4);
  float4 f2 = *(const float4*)(orow + 16 + hi * 8);
  float4 f3 = *(const float4*)(orow + 16 + hi * 8 + 4);
  v16h a;
  a[0]=(_Float16)f0.x; a[1]=(_Float16)f0.y; a[2]=(_Float16)f0.z; a[3]=(_Float16)f0.w;
  a[4]=(_Float16)f1.x; a[5]=(_Float16)f1.y; a[6]=(_Float16)f1.z; a[7]=(_Float16)f1.w;
  a[8]=(_Float16)f2.x; a[9]=(_Float16)f2.y; a[10]=(_Float16)f2.z; a[11]=(_Float16)f2.w;
  a[12]=(_Float16)f3.x; a[13]=(_Float16)f3.y; a[14]=(_Float16)f3.z; a[15]=(_Float16)f3.w;
  return a;
}

// D layout (ISA 7.12.2, f32 C/D 16x16): lane col = L%16, VGPR r -> row r + 8*(L/16)
DEV void store_relu_h(_Float16* __restrict__ buf, int SW, int cb, int lane, const v8f& acc) {
  int col = lane & 15, hi = lane >> 4;
#pragma unroll
  for (int r = 0; r < 8; ++r) buf[(r + hi * 8) * SW + cb + col] = (_Float16)fmaxf(acc[r], 0.f);
}
DEV void store_h(_Float16* __restrict__ buf, int SW, int cb, int lane, const v8f& acc) {
  int col = lane & 15, hi = lane >> 4;
#pragma unroll
  for (int r = 0; r < 8; ++r) buf[(r + hi * 8) * SW + cb + col] = (_Float16)acc[r];
}
DEV void store_out(float* __restrict__ out, size_t rowBase, int cb, int lane, const v8f& acc) {
  int col = lane & 15, hi = lane >> 4;
#pragma unroll
  for (int r = 0; r < 8; ++r) out[(rowBase + r + hi * 8) * OUTW + cb + col] = acc[r];
}

// dual-tile GEMM step: one B-fragment stream reused for two batch tiles
template <int KT>
DEV void gemm2(const _Float16* __restrict__ wf, float bv,
               const _Float16* __restrict__ a0b, const _Float16* __restrict__ a1b, int SW,
               int lane, v8f& acc0, v8f& acc1) {
#pragma unroll
  for (int i = 0; i < 8; ++i) { acc0[i] = bv; acc1[i] = bv; }
#pragma unroll
  for (int k = 0; k < KT; ++k) {
    v16h b = loadB(wf + k * FRAG, lane);
    v16h a0 = loadA(a0b, SW, k * 32, lane);
    acc0 = wmma16(a0, b, acc0);
    v16h a1 = loadA(a1b, SW, k * 32, lane);
    acc1 = wmma16(a1, b, acc1);
  }
}

DEV float sigmf(float x) { return 1.f / (1.f + expf(-x)); }
DEV float softplusf(float x) { return x > 0.f ? x + log1pf(expf(-x)) : log1pf(expf(x)); }

// ---------------- fused network kernel ----------------
// per wave: 2 batch tiles of 16 rows; per tile LDS slice (halves):
//   HM [16x192] @0, S1 [16x128] @3072, S2 [16x128] @5120, HN [16x64] @7168  => 8192
// S1|S2 (contiguous 8KB) doubles as the f32 async landing buffer for messages_in.
__global__ __launch_bounds__(128) void fused_swarm(
    const float* __restrict__ obs, const float* __restrict__ msgs_in,
    const float* __restrict__ h_lstm, const float* __restrict__ c_lstm,
    const _Float16* __restrict__ wsH, const float* __restrict__ wsB,
    const float* __restrict__ maxlv, const float* __restrict__ minlv,
    float* __restrict__ out) {
  extern __shared__ __align__(32) char smem_raw[];
  _Float16* smem = (_Float16*)smem_raw;

  const int lane = threadIdx.x & 31;
  const int wave = threadIdx.x >> 5;
  const int row16 = lane & 15, hi = lane >> 4, col = lane & 15;
  const size_t b0 = (size_t)blockIdx.x * 128 + wave * 32;   // tile0 rows [b0,b0+16), tile1 +16

  _Float16* HM[2]; _Float16* S1[2]; _Float16* S2[2]; _Float16* HN[2];
#pragma unroll
  for (int t = 0; t < 2; ++t) {
    _Float16* base = smem + (wave * 2 + t) * 8192;
    HM[t] = base; S1[t] = base + 3072; S2[t] = base + 5120; HN[t] = base + 7168;
  }

  // warm L2 for the packed-weight blob (global_prefetch_b8)
  __builtin_prefetch(wsH + F_ih + (size_t)threadIdx.x * 256, 0, 1);
  __builtin_prefetch(wsH + F_msg1 + (size_t)threadIdx.x * 128, 0, 1);

  // ---- stage 0: fire async global->LDS staging of messages_in (f32) into S1|S2 ----
  // ASYNCcnt-tracked; overlapped with the obs GEMM below.
#pragma unroll
  for (int t = 0; t < 2; ++t) {
    unsigned int fbase = (unsigned int)(uintptr_t)S1[t];      // LDS byte offset (addr[31:0])
    for (int p = 0; p < 4; ++p) {
      const float* src = msgs_in + ((size_t)p * BATCH + b0 + t * 16 + row16) * 32 + hi * 16;
      unsigned int dst = fbase + (unsigned int)((row16 * 128 + p * 32 + hi * 16) * 4);
#pragma unroll
      for (int q = 0; q < 4; ++q) async_ld_b128(dst + q * 16, src + q * 4);
    }
  }

  // ---- stage 1: x = relu(obs @ W_in^T + b_in)  -> HM cols 0..63 (f16) ----
  v16h aObs[2];
#pragma unroll
  for (int t = 0; t < 2; ++t) aObs[t] = loadA_obs(obs + (b0 + t * 16 + row16) * 32, lane);
  for (int n = 0; n < 4; ++n) {
    float bv = wsB[BI_in + n * 16 + col];
    v8f a0, a1;
#pragma unroll
    for (int i = 0; i < 8; ++i) { a0[i] = bv; a1[i] = bv; }
    v16h b = loadB(wsH + F_Win + n * FRAG, lane);
    a0 = wmma16(aObs[0], b, a0);
    a1 = wmma16(aObs[1], b, a1);
    store_relu_h(HM[0], 192, n * 16, lane, a0);
    store_relu_h(HM[1], 192, n * 16, lane, a1);
  }

  // ---- stage 2: convert staged f32 messages -> f16 HM cols 64..191 (h_mbar = [x | agg]) ----
  wait_async0();
#pragma unroll
  for (int t = 0; t < 2; ++t) {
    const float* fsrc = (const float*)S1[t];
    for (int p = 0; p < 4; ++p) {
      const float* s = fsrc + row16 * 128 + p * 32 + hi * 16;
      _Float16* dst = HM[t] + row16 * 192 + 64 + p * 32 + hi * 16;
#pragma unroll
      for (int q = 0; q < 4; ++q) {
        float4 f = *(const float4*)(s + q * 4);
        v4h h; h[0]=(_Float16)f.x; h[1]=(_Float16)f.y; h[2]=(_Float16)f.z; h[3]=(_Float16)f.w;
        *(v4h*)(dst + q * 4) = h;
      }
    }
  }
  // zero S1 pad columns 112..127 (stride-128 view) now that the f32 buffer is consumed
#pragma unroll
  for (int t = 0; t < 2; ++t) *(v8h*)(S1[t] + row16 * 128 + 112 + hi * 8) = (v8h)0;

  // ---- stage 3: per-port message MLPs; msgs_out straight to global ----
  for (int p = 0; p < 4; ++p) {
    for (int n = 0; n < 7; ++n) {               // m1 = relu(x W1^T + b1) -> S1 (stride 128)
      v8f a0, a1;
      gemm2<2>(wsH + F_msg1 + p * F_msg1_s + n * 2 * FRAG,
               wsB[BI_m1 + p * 112 + n * 16 + col], HM[0], HM[1], 192, lane, a0, a1);
      store_relu_h(S1[0], 128, n * 16, lane, a0);
      store_relu_h(S1[1], 128, n * 16, lane, a1);
    }
    for (int n = 0; n < 2; ++n) {               // msg = m1 W2^T + b2 -> out cols 32+p*32+...
      v8f a0, a1;
      gemm2<4>(wsH + F_msg2 + p * F_msg2_s + n * 4 * FRAG,
               wsB[BI_m2 + p * 32 + n * 16 + col], S1[0], S1[1], 128, lane, a0, a1);
      store_out(out, b0, 32 + p * 32 + n * 16, lane, a0);
      store_out(out, b0 + 16, 32 + p * 32 + n * 16, lane, a1);
    }
  }

  // ---- stage 4: stage h_lstm into HN (f16, cols 50..63 zero) ----
#pragma unroll
  for (int t = 0; t < 2; ++t) {
    const float* hrow = h_lstm + (b0 + t * 16 + row16) * 50;
    _Float16* dst = HN[t] + row16 * 64 + hi * 32;
#pragma unroll
    for (int j = 0; j < 32; ++j) {
      int c = hi * 32 + j;
      dst[j] = (c < 50) ? (_Float16)hrow[c] : (_Float16)0.f;
    }
  }

  // ---- stage 5: LSTM cell; 4 gates as concurrent accumulators ----
  for (int n = 0; n < 4; ++n) {                 // within-gate column tile
    v8f acc[4][2];
#pragma unroll
    for (int q = 0; q < 4; ++q) {
      float bv = wsB[BI_l + q * 64 + n * 16 + col];
#pragma unroll
      for (int i = 0; i < 8; ++i) { acc[q][0][i] = bv; acc[q][1][i] = bv; }
    }
#pragma unroll
    for (int k = 0; k < 6; ++k) {               // h_mbar part (K=192)
      v16h a0 = loadA(HM[0], 192, k * 32, lane);
      v16h a1 = loadA(HM[1], 192, k * 32, lane);
#pragma unroll
      for (int q = 0; q < 4; ++q) {
        v16h b = loadB(wsH + F_ih + ((q * 4 + n) * 6 + k) * FRAG, lane);
        acc[q][0] = wmma16(a0, b, acc[q][0]);
        acc[q][1] = wmma16(a1, b, acc[q][1]);
      }
    }
#pragma unroll
    for (int k = 0; k < 2; ++k) {               // hidden part (K=64 padded)
      v16h a0 = loadA(HN[0], 64, k * 32, lane);
      v16h a1 = loadA(HN[1], 64, k * 32, lane);
#pragma unroll
      for (int q = 0; q < 4; ++q) {
        v16h b = loadB(wsH + F_hh + ((q * 4 + n) * 2 + k) * FRAG, lane);
        acc[q][0] = wmma16(a0, b, acc[q][0]);
        acc[q][1] = wmma16(a1, b, acc[q][1]);
      }
    }
    int j = n * 16 + col;                       // hidden index 0..63; valid < 50
    bool valid = j < 50;
#pragma unroll
    for (int t = 0; t < 2; ++t)
#pragma unroll
      for (int r = 0; r < 8; ++r) {
        size_t gb = b0 + t * 16 + r + hi * 8;
        float iv = sigmf(acc[0][t][r]), fv = sigmf(acc[1][t][r]);
        float gv = tanhf(acc[2][t][r]), ov = sigmf(acc[3][t][r]);
        float co = valid ? c_lstm[gb * 50 + j] : 0.f;
        float cn = fv * co + iv * gv;
        float hn = ov * tanhf(cn);
        if (valid) { out[gb * OUTW + 224 + j] = hn; out[gb * OUTW + 274 + j] = cn; }
        S2[t][(r + hi * 8) * 64 + j] = valid ? (_Float16)hn : (_Float16)0.f;  // h_new view
      }
  }

  // ---- stage 6: update MLP ----
  for (int n = 0; n < 7; ++n) {                 // u1: relu(h_new W_u1^T+b) -> S1
    v8f a0, a1;
    gemm2<2>(wsH + F_u1 + n * 2 * FRAG, wsB[BI_u1 + n * 16 + col], S2[0], S2[1], 64, lane, a0, a1);
    store_relu_h(S1[0], 128, n * 16, lane, a0);
    store_relu_h(S1[1], 128, n * 16, lane, a1);
  }
#pragma unroll
  for (int t = 0; t < 2; ++t)                   // zero S2 pad cols 112..127 before reuse
    *(v8h*)(S2[t] + row16 * 128 + 112 + hi * 8) = (v8h)0;
  for (int n = 0; n < 7; ++n) {                 // u2: relu -> S2
    v8f a0, a1;
    gemm2<4>(wsH + F_u2 + n * 4 * FRAG, wsB[BI_u2 + n * 16 + col], S1[0], S1[1], 128, lane, a0, a1);
    store_relu_h(S2[0], 128, n * 16, lane, a0);
    store_relu_h(S2[1], 128, n * 16, lane, a1);
  }
  for (int n = 0; n < 4; ++n) {                 // u3: h2 -> out cols 160..223 and HN (f16)
    v8f a0, a1;
    gemm2<4>(wsH + F_u3 + n * 4 * FRAG, wsB[BI_u3 + n * 16 + col], S2[0], S2[1], 128, lane, a0, a1);
    store_out(out, b0, 160 + n * 16, lane, a0);
    store_out(out, b0 + 16, 160 + n * 16, lane, a1);
    store_h(HN[0], 64, n * 16, lane, a0);
    store_h(HN[1], 64, n * 16, lane, a1);
  }

  // ---- stage 7: mu / var heads ----
  {
    v8f a0, a1;                                 // mu -> out cols 0..15
    gemm2<2>(wsH + F_mv, wsB[BI_mv + col], HN[0], HN[1], 64, lane, a0, a1);
    store_out(out, b0, 0, lane, a0);
    store_out(out, b0 + 16, 0, lane, a1);
  }
  {
    v8f a0, a1;                                 // logvar -> clamp -> exp -> out cols 16..31
    gemm2<2>(wsH + F_mv + 2 * FRAG, wsB[BI_mv + 16 + col], HN[0], HN[1], 64, lane, a0, a1);
    float mx = maxlv[col], mn = minlv[col];
#pragma unroll
    for (int t = 0; t < 2; ++t) {
      v8f& a = t ? a1 : a0;
#pragma unroll
      for (int r = 0; r < 8; ++r) {
        float lv = a[r];
        lv = mx - softplusf(mx - lv);
        lv = mn + softplusf(lv - mn);
        size_t gb = b0 + t * 16 + r + hi * 8;
        out[gb * OUTW + 16 + col] = expf(lv);
      }
    }
  }
}

// ---------------- host ----------------
extern "C" void kernel_launch(void* const* d_in, const int* in_sizes, int n_in,
                              void* d_out, int out_size, void* d_ws, size_t ws_size,
                              hipStream_t stream) {
  const float* obs     = (const float*)d_in[0];
  const float* msgs_in = (const float*)d_in[1];
  const float* h_lstm  = (const float*)d_in[2];
  const float* c_lstm  = (const float*)d_in[3];
  const float* W_in    = (const float*)d_in[4];
  const float* b_in    = (const float*)d_in[5];
  const float* msg_W1  = (const float*)d_in[6];
  const float* msg_b1  = (const float*)d_in[7];
  const float* msg_W2  = (const float*)d_in[8];
  const float* msg_b2  = (const float*)d_in[9];
  const float* W_ih    = (const float*)d_in[10];
  const float* W_hh    = (const float*)d_in[11];
  const float* b_ih    = (const float*)d_in[12];
  const float* b_hh    = (const float*)d_in[13];
  const float* W_u1    = (const float*)d_in[14];
  const float* b_u1    = (const float*)d_in[15];
  const float* W_u2    = (const float*)d_in[16];
  const float* b_u2    = (const float*)d_in[17];
  const float* W_u3    = (const float*)d_in[18];
  const float* b_u3    = (const float*)d_in[19];
  const float* W_mu    = (const float*)d_in[20];
  const float* b_mu    = (const float*)d_in[21];
  const float* W_var   = (const float*)d_in[22];
  const float* b_var   = (const float*)d_in[23];
  const float* maxlv   = (const float*)d_in[24];
  const float* minlv   = (const float*)d_in[25];
  (void)in_sizes; (void)n_in; (void)out_size; (void)ws_size;

  _Float16* wsH = (_Float16*)d_ws;
  float*    wsB = (float*)((char*)d_ws + (size_t)F_END * 2);

  auto gp = [&](const float* W, int foff, int nT, int kT, int outA, int inA, int gate) {
    int tot = nT * kT * 512;
    pack_frags<<<(tot + 255) / 256, 256, 0, stream>>>(W, wsH + foff, nT, kT, outA, inA, gate);
  };
  gp(W_in, F_Win, 4, 1, 64, 32, 0);
  for (int p = 0; p < 4; ++p) gp(msg_W1 + p * 6400, F_msg1 + p * F_msg1_s, 7, 2, 100, 64, 0);
  for (int p = 0; p < 4; ++p) gp(msg_W2 + p * 3200, F_msg2 + p * F_msg2_s, 2, 4, 32, 100, 0);
  gp(W_ih, F_ih, 16, 6, 0, 192, 1);
  gp(W_hh, F_hh, 16, 2, 0, 50, 1);
  gp(W_u1, F_u1, 7, 2, 100, 50, 0);
  gp(W_u2, F_u2, 7, 4, 100, 100, 0);
  gp(W_u3, F_u3, 4, 4, 64, 100, 0);
  gp(W_mu,  F_mv,            1, 2, 16, 64, 0);
  gp(W_var, F_mv + 2 * FRAG, 1, 2, 16, 64, 0);

  pack_bias<<<1, 64, 0, stream>>>(b_in, wsB + BI_in, 64, 64);
  for (int p = 0; p < 4; ++p)
    pack_bias<<<2, 64, 0, stream>>>(msg_b1 + p * 100, wsB + BI_m1 + p * 112, 100, 112);
  pack_bias<<<2, 64, 0, stream>>>(msg_b2, wsB + BI_m2, 128, 128);
  pack_bias_lstm<<<4, 64, 0, stream>>>(b_ih, b_hh, wsB + BI_l);
  pack_bias<<<2, 64, 0, stream>>>(b_u1, wsB + BI_u1, 100, 112);
  pack_bias<<<2, 64, 0, stream>>>(b_u2, wsB + BI_u2, 100, 112);
  pack_bias<<<1, 64, 0, stream>>>(b_u3, wsB + BI_u3, 64, 64);
  pack_bias<<<1, 64, 0, stream>>>(b_mu,  wsB + BI_mv, 16, 16);
  pack_bias<<<1, 64, 0, stream>>>(b_var, wsB + BI_mv + 16, 16, 16);

  // 4 waves/block, 2 tiles/wave, 16 rows/tile => 128 rows/block
  const int blocks = BATCH / 128;               // 2048
  const size_t shmem = 4 * 2 * 8192 * sizeof(_Float16);  // 128 KB dynamic LDS
  fused_swarm<<<blocks, 128, shmem, stream>>>(obs, msgs_in, h_lstm, c_lstm,
                                              wsH, wsB, maxlv, minlv, (float*)d_out);
}
